// Actor_83056077570406
// MI455X (gfx1250) — matrix operand order repaired and checked by reading, not compiled, not run on hardware
//
#include <hip/hip_runtime.h>
#include <hip/hip_bf16.h>

// ---------------- problem constants (match reference) ----------------
#define B_ROWS 65536
#define OBS    512
#define ACT    64
#define NE     16
#define MT     64      // rows per workgroup (4 waves x 16)
#define STR    520     // padded LDS row stride in bf16 elems (1040B, 16B aligned)
#define WHALF  (NE * ACT * OBS)   // elems in one weight tensor (Wm or Ws)

typedef __attribute__((ext_vector_type(16))) __bf16   bf16x16;
typedef __attribute__((ext_vector_type(4)))  __bf16   bf16x4;
typedef __attribute__((ext_vector_type(8)))  float    floatx8;
typedef __attribute__((ext_vector_type(4)))  float    floatx4;
typedef __attribute__((ext_vector_type(2)))  float    floatx2;
typedef __attribute__((ext_vector_type(4)))  unsigned int uintx4;
typedef __attribute__((ext_vector_type(8)))  unsigned int uintx8;
typedef __attribute__((ext_vector_type(8)))  int          intx8;
typedef __attribute__((ext_vector_type(4)))  int          intx4;

static __device__ __forceinline__ floatx8 wmma_bf16(bf16x16 a, bf16x16 b, floatx8 c) {
  // D = A(16x32 bf16) x B(32x16 bf16) + C(16x16 f32)
  return __builtin_amdgcn_wmma_f32_16x16x32_bf16(
      /*neg_a=*/false, a, /*neg_b=*/false, b,
      /*c_mod=*/(short)0, c, /*reuse_a=*/false, /*reuse_b=*/false);
}

// Load a 16-value bf16 lane fragment as two ds_load_b128 (chunk at p, chunk at p+delta elems).
static __device__ __forceinline__ bf16x16 load_frag(const __bf16* p, int delta) {
  uintx4 lo = *(const uintx4*)(p);
  uintx4 hi = *(const uintx4*)(p + delta);
  uintx8 u = {lo[0], lo[1], lo[2], lo[3], hi[0], hi[1], hi[2], hi[3]};
  return __builtin_bit_cast(bf16x16, u);
}

// TDM: DMA one 64x512 bf16 matrix (row-major, stride 512) from global into LDS,
// inserting 16B of padding after every 1024B row => LDS row stride = 520 elems.
// D# packing per CDNA5 ISA ch.8 (group0 128b, group1 256b; groups 2/3 zero for 2D).
// This toolchain's builtin takes 6 args (incl. an extra int32x8 group before cpol).
static __device__ __forceinline__ void tdm_stage_weights(const __bf16* gsrc, __bf16* ldst) {
  unsigned long long ga = (unsigned long long)(uintptr_t)gsrc;   // 57-bit byte address
  unsigned lds = (unsigned)(uintptr_t)(void*)ldst;               // LDS byte offset

  uintx4 g0;
  g0[0] = 1u;                                            // count=1 (valid), user mode
  g0[1] = lds;                                           // lds_addr[31:0]
  g0[2] = (unsigned)(ga & 0xFFFFFFFFu);                  // global_addr[31:0]
  g0[3] = (unsigned)((ga >> 32) & 0x1FFFFFFu)            // global_addr[56:32]
        | (2u << 30);                                    // type=2 ("image")

  intx8 g1;
  g1[0] = (1 << 16)                                      // data_size = 2 bytes
        | (1 << 20)                                      // pad_enable
        | (7 << 22)                                      // pad_interval: 256 dwords (1024B row)
        | (3 << 25);                                     // pad_amount: 4 dwords (16B)
  g1[1] = (OBS & 0xFFFF) << 16;                          // tensor_dim0[15:0]=512 (abar addr=0)
  g1[2] = (ACT & 0xFFFF) << 16;                          // tensor_dim1[15:0]=64
  g1[3] = (OBS & 0xFFFF) << 16;                          // tile_dim0=512
  g1[4] = (ACT & 0xFFFF);                                // tile_dim1=64, tile_dim2=0
  g1[5] = OBS;                                           // tensor_dim0_stride[31:0]=512
  g1[6] = 0;                                             // stride hi / dim1_stride (unused, 2D)
  g1[7] = 0;

  intx4 g2 = {0, 0, 0, 0};
  intx4 g3 = {0, 0, 0, 0};
  intx8 g4 = {0, 0, 0, 0, 0, 0, 0, 0};                   // extra group (unused, zero)

  __builtin_amdgcn_tensor_load_to_lds(g0, g1, g2, g3, g4, /*cpol=*/0);
}

// ---------------- pre-pass: Wm/Ws fp32 -> bf16 into workspace ----------------
__global__ __launch_bounds__(256)
void cvt_weights_bf16(const float* __restrict__ Wm, const float* __restrict__ Ws,
                      __bf16* __restrict__ dst) {
  int i = blockIdx.x * blockDim.x + threadIdx.x;          // one float4 per tensor per thread
  floatx4 f = *(const floatx4*)(Wm + (size_t)i * 4);
  floatx4 g = *(const floatx4*)(Ws + (size_t)i * 4);
  bf16x4 hf, hg;
  hf[0] = (__bf16)f[0]; hf[1] = (__bf16)f[1]; hf[2] = (__bf16)f[2]; hf[3] = (__bf16)f[3];
  hg[0] = (__bf16)g[0]; hg[1] = (__bf16)g[1]; hg[2] = (__bf16)g[2]; hg[3] = (__bf16)g[3];
  *(bf16x4*)(dst + (size_t)i * 4)         = hf;
  *(bf16x4*)(dst + WHALF + (size_t)i * 4) = hg;
}

// ---------------- fused MoE actor kernel ----------------
__global__ __launch_bounds__(128, 1)
void moe_actor_wmma(const float* __restrict__ x,
                    const float* __restrict__ Wg,
                    const float* __restrict__ bg,
                    const __bf16* __restrict__ wpack,   // [Wm_bf16 | Ws_bf16] in d_ws
                    const float* __restrict__ bm,
                    const float* __restrict__ bs,
                    float* __restrict__ out) {
  __shared__ __bf16 xs[MT * STR];     // x tile, bf16, A-layout friendly (row-major padded)
  __shared__ __bf16 wl[MT * STR];     // one expert weight matrix (64x512) bf16 via TDM
  __shared__ float  llog[MT * NE];    // router logits
  __shared__ float  lgate[MT * NE];   // sparse gate (softmax probs, top-2 kept)
  __shared__ int    gidx2[MT * 2];    // top-2 expert indices per row
  __shared__ float  gval2[MT * 2];    // top-2 gate values per row

  const int tid   = threadIdx.x;
  const int lane  = tid & 31;
  const int wave  = tid >> 5;                 // 0..3
  const int n     = lane & 15;                // N index / column lane
  const int hiH   = (lane & 16) ? 1 : 0;      // upper lane half
  const int koff8  = hiH ? 8  : 0;            // unused (A and B fragments use koff16)
  const int koff16 = hiH ? 16 : 0;            // B-fragment chunk offset
  const int hi8   = hiH ? 8 : 0;              // C-layout row offset
  const int row0  = blockIdx.x * MT;          // first global row of this WG
  const int lrow  = (wave << 4) + n;          // local row owned by this lane (A rows)

  // ---------------- stage x tile -> LDS (fp32 -> bf16) ----------------
  for (int it = 0; it < 64; ++it) {
    int idx = tid + (it << 7);
    int r  = idx >> 7;          // 0..63
    int c4 = idx & 127;         // float4 column
    floatx4 f = *(const floatx4*)(x + (size_t)(row0 + r) * OBS + (c4 << 2));
    bf16x4 h;
    h[0] = (__bf16)f[0]; h[1] = (__bf16)f[1];
    h[2] = (__bf16)f[2]; h[3] = (__bf16)f[3];
    *(bf16x4*)(xs + r * STR + (c4 << 2)) = h;
  }
  __syncthreads();

  // ---------------- router logits via WMMA ----------------
  {
    floatx8 c = {};
    for (int s = 0; s < 16; ++s) {
      bf16x16 a = load_frag(xs + lrow * STR + (s << 5) + koff16, 8);
      bf16x16 b;
      #pragma unroll
      for (int v = 0; v < 8; ++v) {
        floatx2 w2 = *(const floatx2*)(Wg + n * OBS + (s << 5) + koff16 + (v << 1));
        b[2 * v]     = (__bf16)w2[0];
        b[2 * v + 1] = (__bf16)w2[1];
      }
      c = wmma_bf16(a, b, c);
    }
    #pragma unroll
    for (int j = 0; j < 8; ++j)
      llog[(((wave << 4) + j + hi8) << 4) + n] = c[j];
  }
  __syncthreads();

  // ---------------- softmax + top-2 -> sparse gate ----------------
  if (tid < MT) {
    float L[NE];
    float mx = -1e30f;
    #pragma unroll
    for (int e = 0; e < NE; ++e) {
      L[e] = llog[(tid << 4) + e] + bg[e];
      mx = fmaxf(mx, L[e]);
    }
    float sum = 0.f;
    #pragma unroll
    for (int e = 0; e < NE; ++e) { L[e] = __expf(L[e] - mx); sum += L[e]; }
    float inv = 1.0f / sum;
    int i0 = 0, i1 = 0; float v0 = -1.f, v1 = -1.f;
    #pragma unroll
    for (int e = 0; e < NE; ++e) {
      float p = L[e] * inv;
      if (p > v0)      { v1 = v0; i1 = i0; v0 = p; i0 = e; }
      else if (p > v1) { v1 = p;  i1 = e; }
    }
    #pragma unroll
    for (int e = 0; e < NE; ++e) lgate[(tid << 4) + e] = 0.0f;
    lgate[(tid << 4) + i0] = v0;
    lgate[(tid << 4) + i1] = v1;
    gidx2[2 * tid] = i0; gidx2[2 * tid + 1] = i1;
    gval2[2 * tid] = v0; gval2[2 * tid + 1] = v1;
  }
  __syncthreads();

  // ---------------- main dense-over-experts WMMA accumulation ----------------
  floatx8 acc[8] = {};   // tiles 0..3: mean cols 0..63 ; tiles 4..7: log_std cols 0..63

  for (int e = 0; e < NE; ++e) {
    float g = lgate[(lrow << 4) + e];
    __bf16 gb = (__bf16)g;
    bf16x16 gs;
    #pragma unroll
    for (int i = 0; i < 16; ++i) gs[i] = gb;

    #pragma unroll
    for (int ph = 0; ph < 2; ++ph) {          // ph0: Wm -> mean, ph1: Ws -> log_std
      __syncthreads();                        // previous users of wl are done
      // TDM: one instruction stages the whole 64x512 bf16 expert matrix with padding
      if (wave == 0) {
        const __bf16* wsrc = wpack + (size_t)ph * WHALF + (size_t)e * ACT * OBS;
        tdm_stage_weights(wsrc, wl);
        __builtin_amdgcn_s_wait_tensorcnt(0);
      }
      __syncthreads();

      // skip compute if none of this wave's 16 rows routed to expert e
      if (__builtin_amdgcn_ballot_w32(g != 0.0f) == 0u) continue;

      for (int s = 0; s < 16; ++s) {
        bf16x16 a = load_frag(xs + lrow * STR + (s << 5) + koff16, 8);
        a = a * gs;                           // fold gate into A: sum_e g_e * x W_e^T
        #pragma unroll
        for (int tt = 0; tt < 4; ++tt) {
          const __bf16* bp = wl + (((tt << 4) + n) * STR) + (s << 5) + koff16;
          bf16x16 b = load_frag(bp, 8);
          acc[ph * 4 + tt] = wmma_bf16(a, b, acc[ph * 4 + tt]);
        }
      }
    }
  }

  // ---------------- epilogue: bias (top-2 gather), tanh-affine, store ----------------
  int   i0a[8], i1a[8];
  float v0a[8], v1a[8];
  #pragma unroll
  for (int j = 0; j < 8; ++j) {
    int r = (wave << 4) + j + hi8;
    i0a[j] = gidx2[2 * r]; i1a[j] = gidx2[2 * r + 1];
    v0a[j] = gval2[2 * r]; v1a[j] = gval2[2 * r + 1];
  }

  #pragma unroll
  for (int tt = 0; tt < 8; ++tt) {
    const float* bp = (tt < 4) ? bm : bs;
    const size_t obase = (tt < 4) ? (size_t)0 : (size_t)B_ROWS * ACT;
    int col = ((tt & 3) << 4) + n;
    #pragma unroll
    for (int j = 0; j < 8; ++j) {
      float v = acc[tt][j]
              + v0a[j] * bp[i0a[j] * ACT + col]
              + v1a[j] * bp[i1a[j] * ACT + col];
      if (tt >= 4) {                       // log_std path
        v = tanhf(v);
        v = 3.5f * v - 1.5f;               // LOG_STD_MIN + 0.5*(MAX-MIN)*(v+1)
      }
      int grow = row0 + (wave << 4) + j + hi8;
      out[obase + (size_t)grow * ACT + col] = v;
    }
  }
}

extern "C" void kernel_launch(void* const* d_in, const int* in_sizes, int n_in,
                              void* d_out, int out_size, void* d_ws, size_t ws_size,
                              hipStream_t stream) {
  (void)in_sizes; (void)n_in; (void)ws_size; (void)out_size;
  const float* x  = (const float*)d_in[0];
  const float* Wg = (const float*)d_in[1];
  const float* bg = (const float*)d_in[2];
  const float* Wm = (const float*)d_in[3];
  const float* bm = (const float*)d_in[4];
  const float* Ws = (const float*)d_in[5];
  const float* bs = (const float*)d_in[6];
  // d_in[7] = training flag (eval path only)
  float* out = (float*)d_out;
  __bf16* wpack = (__bf16*)d_ws;   // 2 MB: [Wm bf16 (512K elems) | Ws bf16 (512K elems)]

  // pre-pass: weights fp32 -> bf16 (131072 threads, one float4 per tensor each)
  cvt_weights_bf16<<<dim3(512), dim3(256), 0, stream>>>(Wm, Ws, wpack);

  dim3 grid(B_ROWS / MT);   // 1024
  dim3 block(128);          // 4 wave32
  moe_actor_wmma<<<grid, block, 0, stream>>>(x, Wg, bg, wpack, bm, bs, out);
}
